// RNN_64364379898566
// MI455X (gfx1250) — matrix operand order repaired and checked by reading, not compiled, not run on hardware
//
#include <hip/hip_runtime.h>
#include <math.h>

// ---------------------------------------------------------------------------
// RNN LM: emb gather -> (only last block!) precomputed input proj ->
// sequential tanh recurrence (persistent kernel, W_hh^T resident in LDS) ->
// big bf16 WMMA head GEMM (B*T x C) @ (C x V).
// Head/pre GEMM: 128x128 WG tile, 8 waves of 32x64, double-buffered LDS,
// one barrier per K-step, software-pipelined global->LDS staging.
// ---------------------------------------------------------------------------

#define B_  8
#define T_  512
#define C_  1024
#define V_  32000
#define NWG_REC 16          // persistent workgroups for recurrence
#define SAP 40              // LDS row pad (elems): 80B rows, 16B aligned, bank-spread
#define SWS 1032            // LDS row stride for resident W_hh slice (2064B = 16*129)
#define LBUF (128 * SAP)    // one LDS buffer (elems)

typedef __attribute__((ext_vector_type(16))) __bf16 v16bf;
typedef __attribute__((ext_vector_type(8)))  __bf16 v8bf;
typedef __attribute__((ext_vector_type(8)))  float  v8f;

// ---- WMMA fragment loaders (per CDNA5 ISA 7.12.2 16-bit layouts) -----------
// A (16xK tile, row-major): lane L(0..15): row M=L, K=0..7 & 16..23;
//                           lane L+16:    row M=L, K=8..15 & 24..31
__device__ __forceinline__ v16bf frag_a(const __bf16* tile, int stride) {
  int lane = threadIdx.x & 31;
  const __bf16* p = tile + (lane & 15) * stride + ((lane >> 4) << 3);
  v8bf lo = *(const v8bf*)(p);
  v8bf hi = *(const v8bf*)(p + 16);
  return __builtin_shufflevector(lo, hi, 0,1,2,3,4,5,6,7,8,9,10,11,12,13,14,15);
}
// B (KxN tile from N-major storage): lane L(0..15): col N=L, K=0..7 & 16..23;
//                                    lane L+16: col N=L, K=8..15 & 24..31
__device__ __forceinline__ v16bf frag_b(const __bf16* tile, int stride) {
  int lane = threadIdx.x & 31;
  const __bf16* p = tile + (lane & 15) * stride + ((lane >> 4) << 3);
  v8bf lo = *(const v8bf*)(p);
  v8bf hi = *(const v8bf*)(p + 16);
  return __builtin_shufflevector(lo, hi, 0,1,2,3,4,5,6,7,8,9,10,11,12,13,14,15);
}

__device__ __forceinline__ v8f wmma_bf16(v16bf a, v16bf b, v8f c) {
  return __builtin_amdgcn_wmma_f32_16x16x32_bf16(false, a, false, b, (short)0, c,
                                                 false, false);
}

// ---- shared 128x128 GEMM core: C128x128 += A(128xK) * B^T(128xK) -----------
// 256 thr / 8 waves; wave (mw = w&3, nw = w>>2) computes rows 32mw..+32,
// cols 64nw..+64 => acc[2][4]. Double-buffered LDS, one barrier per K-step.
__device__ __forceinline__ void gemm_core_128(const __bf16* __restrict__ Ag0,
                                              const __bf16* __restrict__ Bg0,
                                              __bf16* sA, __bf16* sB,
                                              v8f (&acc)[2][4]) {
  int tid = threadIdx.x;
  int wave = tid >> 5;
  int mw = wave & 3, nw = wave >> 2;
  int row = tid >> 1, off = (tid & 1) * 16;       // 128 rows x 32 k, 16 elem/thr
  const __bf16* ag = Ag0 + (size_t)row * C_ + off;
  const __bf16* bg = Bg0 + (size_t)row * C_ + off;
  int sidx = row * SAP + off;

  // prologue: stage chunk 0 into buffer 0
  v8bf ra0 = *(const v8bf*)(ag);
  v8bf ra1 = *(const v8bf*)(ag + 8);
  v8bf rb0 = *(const v8bf*)(bg);
  v8bf rb1 = *(const v8bf*)(bg + 8);
  *(v8bf*)(sA + sidx) = ra0; *(v8bf*)(sA + sidx + 8) = ra1;
  *(v8bf*)(sB + sidx) = rb0; *(v8bf*)(sB + sidx + 8) = rb1;
  __syncthreads();

  const int NK = C_ / 32;
  for (int kk = 0; kk < NK; ++kk) {
    int cb = kk & 1;
    const __bf16* cA = sA + cb * LBUF;
    const __bf16* cB = sB + cb * LBUF;
    // issue next chunk's global loads before the WMMA block
    if (kk + 1 < NK) {
      ra0 = *(const v8bf*)(ag + (kk + 1) * 32);
      ra1 = *(const v8bf*)(ag + (kk + 1) * 32 + 8);
      rb0 = *(const v8bf*)(bg + (kk + 1) * 32);
      rb1 = *(const v8bf*)(bg + (kk + 1) * 32 + 8);
    }
    v16bf af0 = frag_a(cA + (mw * 32) * SAP, SAP);
    v16bf af1 = frag_a(cA + (mw * 32 + 16) * SAP, SAP);
    for (int f = 0; f < 4; ++f) {
      v16bf bf = frag_b(cB + (nw * 64 + f * 16) * SAP, SAP);
      acc[0][f] = wmma_bf16(af0, bf, acc[0][f]);
      acc[1][f] = wmma_bf16(af1, bf, acc[1][f]);
    }
    if (kk + 1 < NK) {
      __bf16* nA = sA + (cb ^ 1) * LBUF;
      __bf16* nB = sB + (cb ^ 1) * LBUF;
      *(v8bf*)(nA + sidx) = ra0; *(v8bf*)(nA + sidx + 8) = ra1;
      *(v8bf*)(nB + sidx) = rb0; *(v8bf*)(nB + sidx + 8) = rb1;
    }
    __syncthreads();
  }
}

// ---- init: barrier counter + zero hidden-state ping-pong buffers -----------
__global__ void k_init(unsigned* bar, __bf16* hbuf) {
  if (threadIdx.x == 0) bar[0] = 0u;
  for (int i = threadIdx.x; i < 2 * 16 * C_; i += blockDim.x)
    hbuf[i] = (__bf16)0.0f;
}

// ---- embedding gather + fp32->bf16, rows stored as r = t*B + b -------------
__global__ void k_embed(const int* __restrict__ idx, const float* __restrict__ emb,
                        __bf16* __restrict__ xbf) {
  int r = blockIdx.x;               // r = t*B + b
  int t = r >> 3, b = r & 7;
  int tok = idx[b * T_ + t];
  const float4* src = (const float4*)(emb + (size_t)tok * C_);
  float4 v = src[threadIdx.x];
  __bf16* dst = xbf + (size_t)r * C_ + threadIdx.x * 4;
  dst[0] = (__bf16)v.x; dst[1] = (__bf16)v.y;
  dst[2] = (__bf16)v.z; dst[3] = (__bf16)v.w;
}

// ---- transpose+cast the two 1024x1024 block-1 weights to N-major bf16 ------
__global__ void k_cast_sq(const float* __restrict__ Wih, const float* __restrict__ Whh,
                          __bf16* __restrict__ WihT, __bf16* __restrict__ WhhT) {
  int id = blockIdx.x * blockDim.x + threadIdx.x;     // 0 .. 2*2^20
  int which = id >> 20;
  int e = id & ((1 << 20) - 1);
  int n = e >> 10, k = e & 1023;
  const float* W = which ? Whh : Wih;
  __bf16* WT = which ? WhhT : WihT;
  WT[(size_t)n * C_ + k] = (__bf16)W[(size_t)k * C_ + n];
}

// ---- transpose+cast W_head (C x V) -> (V x C) bf16 -------------------------
__global__ void k_cast_head(const float* __restrict__ Wh, __bf16* __restrict__ WT) {
  size_t id = (size_t)blockIdx.x * blockDim.x + threadIdx.x;   // V*C
  int v = (int)(id >> 10);
  int c = (int)(id & 1023);
  WT[id] = (__bf16)Wh[(size_t)c * V_ + v];
}

// ---- G = X @ W_ih[1] + (b_ih[1]+b_hh[1]); M=4096 N=1024 K=1024, fp32 out ---
__global__ void k_gemm_pre(const __bf16* __restrict__ Xbf, const __bf16* __restrict__ WihT,
                           const float* __restrict__ b_ih, const float* __restrict__ b_hh,
                           float* __restrict__ G) {
  __shared__ __bf16 sA[2 * LBUF];
  __shared__ __bf16 sB[2 * LBUF];
  int mblk = blockIdx.x, nblk = blockIdx.y;
  v8f acc[2][4];
  v8f zero = {0.f,0.f,0.f,0.f,0.f,0.f,0.f,0.f};
  for (int a = 0; a < 2; ++a) for (int f = 0; f < 4; ++f) acc[a][f] = zero;

  gemm_core_128(Xbf + (size_t)(mblk * 128) * C_,
                WihT + (size_t)(nblk * 128) * C_, sA, sB, acc);

  int tid = threadIdx.x, wave = tid >> 5, lane = tid & 31;
  int mw = wave & 3, nw = wave >> 2;
  int ln = lane & 15, lh = lane >> 4;
  for (int a = 0; a < 2; ++a) {
    for (int f = 0; f < 4; ++f) {
      int col = nblk * 128 + nw * 64 + f * 16 + ln;
      float bias = b_ih[C_ + col] + b_hh[C_ + col];   // block 1
      int rowbase = mblk * 128 + mw * 32 + a * 16 + 8 * lh;
      for (int i = 0; i < 8; ++i)
        G[(size_t)(rowbase + i) * C_ + col] = acc[a][f][i] + bias;
    }
  }
}

// ---- sequential recurrence: persistent, W_hh^T slice resident in LDS -------
// 16 WGs x 128 thr (4 waves). WG wg owns cols [64wg,64wg+64); wave w -> 16 cols.
// h kept as 16 x C bf16 (rows 8..15 stay zero), ping-pong in global.
__global__ void k_recur(const __bf16* __restrict__ WhhT, const float* __restrict__ G,
                        __bf16* __restrict__ hbuf, __bf16* __restrict__ Hout,
                        unsigned* __restrict__ bar) {
  __shared__ __bf16 sW[64 * SWS];    // ~129 KB of the 320 KB CDNA5 LDS
  int wg = blockIdx.x;
  int tid = threadIdx.x, wave = tid >> 5, lane = tid & 31;

  // preload this WG's 64 x 1024 W_hh^T slice into LDS (v8bf copies)
  {
    const __bf16* src = WhhT + (size_t)(wg * 64) * C_;
    for (int e = tid * 8; e < 64 * C_; e += 128 * 8) {
      int n = e >> 10, k = e & 1023;
      *(v8bf*)(sW + n * SWS + k) = *(const v8bf*)(src + (size_t)n * C_ + k);
    }
  }
  __syncthreads();

  __bf16* h0 = hbuf;
  __bf16* h1 = hbuf + 16 * C_;
  const __bf16* sWt = sW + (wave * 16) * SWS;
  int ln = lane & 15, lh = lane >> 4;
  int col = wg * 64 + wave * 16 + ln;

  for (int t = 0; t < T_; ++t) {
    const __bf16* cur = (t & 1) ? h1 : h0;
    __bf16*       nxt = (t & 1) ? h0 : h1;
    v8f acc = {0.f,0.f,0.f,0.f,0.f,0.f,0.f,0.f};
    for (int kk = 0; kk < C_; kk += 32) {
      v16bf af = frag_a(cur + kk, C_);
      v16bf bf = frag_b(sWt + kk, SWS);
      acc = wmma_bf16(af, bf, acc);
    }
    if (lh == 0) {                               // rows 0..7 are the real batch
      for (int i = 0; i < 8; ++i) {
        float pre = acc[i] + G[(size_t)(t * B_ + i) * C_ + col];
        __bf16 hb = (__bf16)tanhf(pre);
        nxt[i * C_ + col] = hb;
        Hout[(size_t)(t * B_ + i) * C_ + col] = hb;
      }
    }
    // device-wide step barrier (monotonic counter: no reset needed, 512*16<2^32)
    __threadfence();
    __syncthreads();
    if (tid == 0) {
      atomicAdd(bar, 1u);
      unsigned target = (unsigned)(t + 1) * NWG_REC;
      while (__hip_atomic_load(bar, __ATOMIC_RELAXED, __HIP_MEMORY_SCOPE_AGENT) < target)
        __builtin_amdgcn_s_sleep(1);
    }
    __syncthreads();
    __threadfence();
  }
}

// ---- head: (4096 x 1024) @ (1024 x 32000) + b_head, row remap to (B,T,V) ---
__global__ void k_head(const __bf16* __restrict__ Hb, const __bf16* __restrict__ WheadT,
                       const float* __restrict__ b_head, float* __restrict__ out) {
  __shared__ __bf16 sA[2 * LBUF];
  __shared__ __bf16 sB[2 * LBUF];
  int nblk = blockIdx.x;             // 250
  int mblk = blockIdx.y;             // 32
  v8f acc[2][4];
  v8f zero = {0.f,0.f,0.f,0.f,0.f,0.f,0.f,0.f};
  for (int a = 0; a < 2; ++a) for (int f = 0; f < 4; ++f) acc[a][f] = zero;

  gemm_core_128(Hb + (size_t)(mblk * 128) * C_,
                WheadT + (size_t)(nblk * 128) * C_, sA, sB, acc);

  int tid = threadIdx.x, wave = tid >> 5, lane = tid & 31;
  int mw = wave & 3, nw = wave >> 2;
  int ln = lane & 15, lh = lane >> 4;
  for (int a = 0; a < 2; ++a) {
    for (int f = 0; f < 4; ++f) {
      int col = nblk * 128 + nw * 64 + f * 16 + ln;
      float bh = b_head[col];
      int rowbase = mblk * 128 + mw * 32 + a * 16 + 8 * lh;
      for (int i = 0; i < 8; ++i) {
        int row = rowbase + i;                   // row = t*B + b
        int t = row >> 3, b = row & 7;
        out[((size_t)b * T_ + t) * V_ + col] = acc[a][f][i] + bh;
      }
    }
  }
}

// ---------------------------------------------------------------------------
extern "C" void kernel_launch(void* const* d_in, const int* in_sizes, int n_in,
                              void* d_out, int out_size, void* d_ws, size_t ws_size,
                              hipStream_t stream) {
  const int*   idx    = (const int*)  d_in[0];
  const float* emb    = (const float*)d_in[1];
  const float* W_ih   = (const float*)d_in[2];   // (2, C, C)
  const float* W_hh   = (const float*)d_in[3];   // (2, C, C)
  const float* b_ih   = (const float*)d_in[4];   // (2, C)
  const float* b_hh   = (const float*)d_in[5];   // (2, C)
  const float* W_head = (const float*)d_in[6];   // (C, V)
  const float* b_head = (const float*)d_in[7];   // (V)
  float* out = (float*)d_out;

  // workspace carve-up (~104 MB)
  char* w = (char*)d_ws;
  __bf16* xbf    = (__bf16*)w; w += (size_t)4096 * C_ * 2;      // 8 MB
  __bf16* WihT   = (__bf16*)w; w += (size_t)C_ * C_ * 2;        // 2 MB
  __bf16* WhhT   = (__bf16*)w; w += (size_t)C_ * C_ * 2;        // 2 MB
  __bf16* WheadT = (__bf16*)w; w += (size_t)V_ * C_ * 2;        // 65.5 MB
  float*  G      = (float*) w; w += (size_t)4096 * C_ * 4;      // 16.8 MB
  __bf16* Hout   = (__bf16*)w; w += (size_t)4096 * C_ * 2;      // 8 MB
  __bf16* hbuf   = (__bf16*)w; w += (size_t)2 * 16 * C_ * 2;    // 64 KB
  unsigned* bar  = (unsigned*)w;

  k_init<<<1, 256, 0, stream>>>(bar, hbuf);
  k_embed<<<4096, 256, 0, stream>>>(idx, emb, xbf);
  k_cast_sq<<<(2 * 1024 * 1024) / 256, 256, 0, stream>>>(
      W_ih + (size_t)C_ * C_, W_hh + (size_t)C_ * C_, WihT, WhhT);
  k_cast_head<<<(V_ * C_) / 256, 256, 0, stream>>>(W_head, WheadT);
  k_gemm_pre<<<dim3(32, 8), 256, 0, stream>>>(xbf, WihT, b_ih, b_hh, G);
  k_recur<<<NWG_REC, 128, 0, stream>>>(WhhT, G, hbuf, Hout, bar);
  k_head<<<dim3(250, 32), 256, 0, stream>>>(Hout, WheadT, b_head, out);
}
